// ColourHistogram_566935683074
// MI455X (gfx1250) — matrix operand order repaired and checked by reading, not compiled, not run on hardware
//
#include <hip/hip_runtime.h>

#define BINS 32
#define IMG_HW (512 * 512)
#define BLOCKS_PER_IMG 64
// -0.5 / sigma^2 * log2(e), sigma = 0.05 ->  exp(-0.5 t^2/s^2) = exp2(KEXP * t^2)
#define KEXP (-288.53900817779268f)

typedef __attribute__((ext_vector_type(16))) _Float16 v16h;
typedef __attribute__((ext_vector_type(8)))  float    v8f;

__global__ void ColourHistogram_zero(float* out, int n) {
    int i = blockIdx.x * blockDim.x + threadIdx.x;
    if (i < n) out[i] = 0.0f;
}

// exp2(KEXP*x^2 + c1*x + c2) with xx = x*x precomputed (shared across 2 bins)
__device__ __forceinline__ _Float16 gaussq(float x, float xx, float c1, float c2) {
    return (_Float16)__builtin_amdgcn_exp2f(fmaf(KEXP, xx, fmaf(c1, x, c2)));
}

__global__ __launch_bounds__(256) void ColourHistogram_wmma(
        const float* __restrict__ image, float* __restrict__ out) {
    __shared__ float smem[BINS * BINS];

    const int img  = blockIdx.y;
    const int lane = threadIdx.x & 31;
    const int wave = threadIdx.x >> 5;
    const int h    = lane >> 4;     // wave half: selects K sub-range in A/B layouts
    const int m    = lane & 15;     // row (A) / column (B/C/D) index within a 16-tile

    for (int i = threadIdx.x; i < BINS * BINS; i += blockDim.x) smem[i] = 0.0f;
    __syncthreads();

    const float* chA = image + (size_t)img * 2 * IMG_HW;
    const float* chB = chA + IMG_HW;

    const int pixPerBlock = IMG_HW / BLOCKS_PER_IMG;   // 4096
    const int pixPerWave  = pixPerBlock / 8;           // 512
    const int p0 = blockIdx.x * pixPerBlock + wave * pixPerWave;

    const float bscale = 1.0f / (BINS - 1);
    const float b0v = (float)m * bscale;          // bin value for tile row/col 0..15
    const float b1v = (float)(m + 16) * bscale;   // bin value for tile row/col 16..31
    // per-lane quadratic coefficients: KEXP*(x-b)^2 = KEXP*x^2 + c1*x + c2
    const float c1_0 = -2.0f * KEXP * b0v, c2_0 = KEXP * b0v * b0v;
    const float c1_1 = -2.0f * KEXP * b1v, c2_1 = KEXP * b1v * b1v;

    v8f acc00 = {}, acc01 = {}, acc10 = {}, acc11 = {};

    for (int p = p0; p < p0 + pixPerWave; p += 64) {
        // lane holds pixels p+2*lane, p+2*lane+1  (global_load_b64, coalesced)
        float2 va = *(const float2*)(chA + p + 2 * lane);
        float2 vb = *(const float2*)(chB + p + 2 * lane);

#pragma unroll
        for (int c = 0; c < 2; ++c) {            // two 32-pixel K-chunks per load
            v16h a0 = {}, a1 = {}, bb0 = {}, bb1 = {};
#pragma unroll
            for (int j = 0; j < 8; ++j) {
                // A 16x32 f16 layout: lanes 0-15 VGPRj -> K=(j<4?0:16)+2*(j&3)[+1];
                //                     lanes 16-31 add +8
                int kA = ((j & 4) << 2) + ((j & 3) << 1) + (h << 3);
                // B 32x16 f16 layout: same K order as A
                int kB = ((j & 4) << 2) + ((j & 3) << 1) + (h << 3);
                // pixel pair (k, k+1) of chunk c lives in lane c*16 + k/2 (comp x/y)
                int sA = (c << 4) + (kA >> 1);
                int sB = (c << 4) + (kB >> 1);

                float xa0 = __shfl(va.x, sA, 32);   // ds_bpermute_b32
                float xa1 = __shfl(va.y, sA, 32);
                float xb0 = __shfl(vb.x, sB, 32);
                float xb1 = __shfl(vb.y, sB, 32);

                float xx;
                xx = xa0 * xa0;
                a0[2 * j]      = gaussq(xa0, xx, c1_0, c2_0);
                a1[2 * j]      = gaussq(xa0, xx, c1_1, c2_1);
                xx = xa1 * xa1;
                a0[2 * j + 1]  = gaussq(xa1, xx, c1_0, c2_0);
                a1[2 * j + 1]  = gaussq(xa1, xx, c1_1, c2_1);
                xx = xb0 * xb0;
                bb0[2 * j]     = gaussq(xb0, xx, c1_0, c2_0);
                bb1[2 * j]     = gaussq(xb0, xx, c1_1, c2_1);
                xx = xb1 * xb1;
                bb0[2 * j + 1] = gaussq(xb1, xx, c1_0, c2_0);
                bb1[2 * j + 1] = gaussq(xb1, xx, c1_1, c2_1);
            }
            // hist[m,n] += sum_k A[m,k]*B[k,n] over this 32-pixel chunk
            acc00 = __builtin_amdgcn_wmma_f32_16x16x32_f16(false, a0, false, bb0,
                                                           (short)0, acc00, false, false);
            acc01 = __builtin_amdgcn_wmma_f32_16x16x32_f16(false, a0, false, bb1,
                                                           (short)0, acc01, false, false);
            acc10 = __builtin_amdgcn_wmma_f32_16x16x32_f16(false, a1, false, bb0,
                                                           (short)0, acc10, false, false);
            acc11 = __builtin_amdgcn_wmma_f32_16x16x32_f16(false, a1, false, bb1,
                                                           (short)0, acc11, false, false);
        }
    }

    // C/D layout: lanes 0-15 -> N=lane, VGPR r -> M=r; lanes 16-31 -> N=lane-16, M=r+8
    const int row = h * 8;
#pragma unroll
    for (int r = 0; r < 8; ++r) {
        atomicAdd(&smem[(row + r) * BINS + m],             acc00[r]);  // ds_add_f32
        atomicAdd(&smem[(row + r) * BINS + 16 + m],        acc01[r]);
        atomicAdd(&smem[(16 + row + r) * BINS + m],        acc10[r]);
        atomicAdd(&smem[(16 + row + r) * BINS + 16 + m],   acc11[r]);
    }
    __syncthreads();

    float* o = out + (size_t)img * BINS * BINS;
    for (int i = threadIdx.x; i < BINS * BINS; i += blockDim.x)
        atomicAdd(&o[i], smem[i]);   // global_atomic_add_f32
}

extern "C" void kernel_launch(void* const* d_in, const int* in_sizes, int n_in,
                              void* d_out, int out_size, void* d_ws, size_t ws_size,
                              hipStream_t stream) {
    const float* image = (const float*)d_in[0];
    float* out = (float*)d_out;

    const int N = in_sizes[0] / (2 * IMG_HW);   // 16 images

    ColourHistogram_zero<<<(out_size + 255) / 256, 256, 0, stream>>>(out, out_size);

    dim3 grid(BLOCKS_PER_IMG, N);               // 64 x 16 = 1024 blocks, 8 waves each
    ColourHistogram_wmma<<<grid, 256, 0, stream>>>(image, out);
}
